// ScaledDotProductAttention_45071386804682
// MI455X (gfx1250) — matrix operand order repaired and checked by reading, not compiled, not run on hardware
//
#include <hip/hip_runtime.h>
#include <hip/hip_bf16.h>

// ---------------------------------------------------------------------------
// Fused scaled-dot-product attention forward for MI455X (gfx1250, wave32).
//   out   [B,H,S,D]  = softmax(QK^T*scale + mask) V
//   attn  [B,H,S,S]  = softmax(QK^T*scale + mask)
// B=2 H=16 S=2048 D=64, fp32 in/out.  Matmuls: v_wmma_f32_16x16x32_bf16.
// One workgroup (8 wave32s) per 16-row Q block; full score row in LDS.
// Weights stream to HBM via GLOBAL_STORE_ASYNC_FROM_LDS_B128 (ASYNCcnt DMA),
// overlapped with the PV WMMA loop.
// ---------------------------------------------------------------------------

typedef __attribute__((ext_vector_type(16))) __bf16 v16bf;
typedef __attribute__((ext_vector_type(8)))  float  v8f;
typedef __attribute__((__vector_size__(16))) int    v4i;    // int4 payload type

#define B_      2
#define H_      16
#define S_      2048
#define D_      64
#define SCALE   0.125f          // 1/sqrt(64)
#define SROW    2052            // 2048 + 4 floats pad; row stride 8208B (16B aligned)
#define NWAVE   8
#define KCHUNK  (S_ / NWAVE)    // 256 k-columns per wave

static __device__ __forceinline__ __bf16 f2bf(float f) {
    return static_cast<__bf16>(f);   // native v_cvt_*bf16_f32 on gfx1250
}

static __device__ __forceinline__ v8f wmma_bf16(v16bf a, v16bf b, v8f c) {
    return __builtin_amdgcn_wmma_f32_16x16x32_bf16(
        /*neg_a=*/false, a, /*neg_b=*/false, b,
        /*c_mod=*/(short)0, c, /*reuse_a=*/false, /*reuse_b=*/false);
}

__global__ __launch_bounds__(256)
void attn_fwd_kernel(const float* __restrict__ Q,
                     const float* __restrict__ K,
                     const float* __restrict__ V,
                     const unsigned char* __restrict__ mask,  // [B,1,1,S], nonzero = masked
                     float* __restrict__ out,                 // [B,H,S,D]
                     float* __restrict__ attn)                // [B,H,S,S]
{
    extern __shared__ char smem_raw[];
    float* s_scores = (float*)smem_raw;                               // 16 x SROW fp32
    float* s_oacc   = (float*)(smem_raw + 16 * SROW * sizeof(float)); // 16 x 64  fp32

    const int tid   = threadIdx.x;
    const int lane  = tid & 31;
    const int wid   = tid >> 5;              // wave id 0..7
    const int qbase = blockIdx.x * 16;       // q-row block
    const int bh    = blockIdx.z * H_ + blockIdx.y;
    const int lhalf = lane >> 4;             // 0: lanes 0-15, 1: lanes 16-31
    const int llow  = lane & 15;

    const unsigned char* mrow = mask + (size_t)blockIdx.z * S_;

    // zero the cross-wave O accumulator (consumed only after 2 barriers)
    for (int i = tid; i < 16 * D_; i += 256) s_oacc[i] = 0.0f;

    // ---------------- load Q A-fragments (16x64 -> two 16x32 bf16 frags) ----
    // A 16x32 layout: lane holds row m=lane&15; element j:
    //   k = (lane>=16 ? 16 : 0) + j
    v16bf aq0, aq1;
    {
        const float* Qrow = Q + ((size_t)bh * S_ + qbase + llow) * D_;
#pragma unroll
        for (int j = 0; j < 16; j += 2) {
            const int p = j >> 1;
            const int d = lhalf * 16 + j;
            float2 q0 = *(const float2*)(Qrow + d);
            float2 q1 = *(const float2*)(Qrow + 32 + d);
            aq0[j] = f2bf(q0.x); aq0[j + 1] = f2bf(q0.y);
            aq1[j] = f2bf(q1.x); aq1[j + 1] = f2bf(q1.y);
        }
    }

    // ---------------- S = Q K^T * scale  (+mask) into LDS -------------------
    // B 32x16 layout: lane holds col n=lane&15; element j: k = (lane>=16?16:0)+j
    const int kstart = wid * KCHUNK;
    for (int t = 0; t < KCHUNK / 16; ++t) {
        const int kb = kstart + t * 16;
        const float* Krow = K + ((size_t)bh * S_ + kb + llow) * D_ + lhalf * 16;
        __builtin_prefetch(Krow + 16 * D_, 0, 3);   // next K tile -> global_prefetch_b8
        v16bf bk0, bk1;
#pragma unroll
        for (int j = 0; j < 16; j += 4) {
            float4 x0 = *(const float4*)(Krow + j);
            float4 x1 = *(const float4*)(Krow + 32 + j);
            bk0[j] = f2bf(x0.x); bk0[j+1] = f2bf(x0.y); bk0[j+2] = f2bf(x0.z); bk0[j+3] = f2bf(x0.w);
            bk1[j] = f2bf(x1.x); bk1[j+1] = f2bf(x1.y); bk1[j+2] = f2bf(x1.z); bk1[j+3] = f2bf(x1.w);
        }
        v8f c = {};
        c = wmma_bf16(aq0, bk0, c);      // d = 0..31
        c = wmma_bf16(aq1, bk1, c);      // d = 32..63
        // C layout: VGPR r -> row M = r + 8*lhalf, col N = llow
        const int col    = kb + llow;
        const bool maskd = (mrow[col] != 0);
#pragma unroll
        for (int r = 0; r < 8; ++r) {
            const int M = r + lhalf * 8;
            s_scores[M * SROW + col] = maskd ? -__builtin_inff() : c[r] * SCALE;
        }
    }
    __syncthreads();

    // ---------------- softmax over full rows (2 rows per wave) --------------
#pragma unroll
    for (int rr = 0; rr < 2; ++rr) {
        const int row = wid * 2 + rr;
        float* Srow = s_scores + row * SROW;
        float mx = -__builtin_inff();
        for (int c = lane; c < S_; c += 32) mx = fmaxf(mx, Srow[c]);
#pragma unroll
        for (int o = 16; o; o >>= 1) mx = fmaxf(mx, __shfl_xor(mx, o, 32));
        float sum = 0.0f;
        for (int c = lane; c < S_; c += 32) {
            float e = __expf(Srow[c] - mx);   // exp(-inf)=0 handles masked cols
            Srow[c] = e;
            sum += e;
        }
#pragma unroll
        for (int o = 16; o; o >>= 1) sum += __shfl_xor(sum, o, 32);
        const float inv = 1.0f / sum;
        for (int c = lane; c < S_; c += 32) Srow[c] *= inv;
    }
    __syncthreads();

    // ---------------- stream attention weights to global --------------------
    // Normalized P lives in LDS: fire the 512MB-wide store stream FIRST so the
    // async DMA (or plain stores) overlaps with the PV WMMA loop below.
    {
        float* wrow = attn + ((size_t)bh * S_ + qbase) * S_;
#if __has_builtin(__builtin_amdgcn_global_store_async_from_lds_b128)
        // LDS -> HBM without touching VGPR data; tracked on ASYNCcnt.
        for (int chunk = tid; chunk < 16 * (S_ / 4); chunk += 256) {
            const int r  = chunk >> 9;          // row 0..15
            const int c4 = chunk & 511;         // 16B chunk within row
            __builtin_amdgcn_global_store_async_from_lds_b128(
                (__attribute__((address_space(1))) v4i*)((v4i*)(wrow + (size_t)r * S_ + c4 * 4)),
                (__attribute__((address_space(3))) v4i*)((v4i*)(s_scores + r * SROW + c4 * 4)),
                0, 0);
        }
#else
        for (int idx = tid; idx < 16 * S_; idx += 256) {
            const int r = idx >> 11;
            const int c = idx & (S_ - 1);
            wrow[(size_t)r * S_ + c] = s_scores[r * SROW + c];
        }
#endif
    }

    // ---------------- O = P V  (per-wave partial, LDS atomic combine) --------
    v8f acc[4] = {v8f{}, v8f{}, v8f{}, v8f{}};
    for (int kk = kstart; kk < kstart + KCHUNK; kk += 32) {
        // A fragment: P[qrow, kk..kk+31] bf16 (same 16x32 A layout, from LDS)
        v16bf ap;
#pragma unroll
        for (int j = 0; j < 16; j += 2) {
            const int p  = j >> 1;
            const int kp = kk + lhalf * 16 + j;
            float2 pv = *(const float2*)(s_scores + llow * SROW + kp);
            ap[j] = f2bf(pv.x); ap[j + 1] = f2bf(pv.y);
        }
        // B fragments: V[kk..kk+31, dtile*16..+15]; element j -> row kk+lhalf*16+j
        const float* Vbase = V + ((size_t)bh * S_ + kk + lhalf * 16) * D_ + llow;
        __builtin_prefetch(Vbase + 32 * D_, 0, 3);
#pragma unroll
        for (int t = 0; t < 4; ++t) {
            v16bf bv;
#pragma unroll
            for (int j = 0; j < 16; ++j) bv[j] = f2bf(Vbase[(size_t)j * D_ + t * 16]);
            acc[t] = wmma_bf16(ap, bv, acc[t]);
        }
    }
    // combine the 8 per-wave partials: ds_add_f32 atomics into s_oacc
#pragma unroll
    for (int t = 0; t < 4; ++t)
#pragma unroll
        for (int r = 0; r < 8; ++r) {
            const int M = r + lhalf * 8;
            atomicAdd(&s_oacc[M * D_ + t * 16 + llow], acc[t][r]);
        }
    __syncthreads();

    // ---------------- final O store -----------------------------------------
    for (int idx = tid; idx < 16 * D_; idx += 256) {
        const int m = idx >> 6;
        const int d = idx & 63;
        out[((size_t)bh * S_ + qbase + m) * D_ + d] = s_oacc[idx];
    }

#if __has_builtin(__builtin_amdgcn_s_wait_asynccnt)
    // drain the async weights stream (S_ENDPGM would also wait, this is explicit)
    __builtin_amdgcn_s_wait_asynccnt(0);
#endif
}

extern "C" void kernel_launch(void* const* d_in, const int* in_sizes, int n_in,
                              void* d_out, int out_size, void* d_ws, size_t ws_size,
                              hipStream_t stream) {
    (void)in_sizes; (void)n_in; (void)out_size; (void)d_ws; (void)ws_size;
    const float*         Q    = (const float*)d_in[0];
    const float*         K    = (const float*)d_in[1];
    const float*         V    = (const float*)d_in[2];
    const unsigned char* mask = (const unsigned char*)d_in[3];  // bool [B,1,1,S]

    float* out  = (float*)d_out;                         // [B,H,S,D]
    float* attn = out + (size_t)B_ * H_ * S_ * D_;       // [B,H,S,S]

    const size_t shmem = (size_t)16 * SROW * sizeof(float)   // scores (~128 KB)
                       + (size_t)16 * D_ * sizeof(float);    // O accumulator

    // allow >64KB dynamic LDS (gfx1250 WGP has 320KB); idempotent, capture-safe
    static_cast<void>(hipFuncSetAttribute(
        (const void*)attn_fwd_kernel,
        hipFuncAttributeMaxDynamicSharedMemorySize, (int)shmem));

    dim3 grid(S_ / 16, H_, B_);   // 128 x 16 x 2 = 4096 workgroups
    dim3 block(256);              // 8 wave32s
    attn_fwd_kernel<<<grid, block, shmem, stream>>>(Q, K, V, mask, out, attn);
}
